// SEWBlock_63737314673250
// MI455X (gfx1250) — matrix operand order repaired and checked by reading, not compiled, not run on hardware
//
#include <hip/hip_runtime.h>
#include <cstdint>

typedef __attribute__((ext_vector_type(16))) _Float16 v16h;
typedef __attribute__((ext_vector_type(8)))  float    v8f;

#define TT  4
#define BB  8
#define CC  128
#define HH  56
#define WW  56
#define HP  58            // padded height (1-px zero halo)
#define WP  58            // padded width
#define HWW (HH*WW)       // 3136 pixels per image
#define PPIX (HP*WP)      // 3364 padded pixels per image
#define NN  (TT*BB)       // 32 images
#define KK  (CC*9)        // 1152 = GEMM K
#define KCN (KK/32)       // 36 K-chunks of 32
#define PTILE 64          // pixels per workgroup

// ---------------------------------------------------------------------------
// Fold BN into conv weights; emit f16 A-matrix fragments pre-swizzled into the
// contiguous-k 16x32 (MxK) A order this kernel uses: lanes 0-15 hold K 0-15,
// lanes 16-31 hold K 16-31.  K ordering: k = (r*3+s)*128 + ci.
// ---------------------------------------------------------------------------
__global__ __launch_bounds__(256) void pack_w_kernel(
    const float* __restrict__ w,      // [Co][Ci][3][3]
    const float* __restrict__ gamma,
    const float* __restrict__ beta,
    const float* __restrict__ mean,
    const float* __restrict__ var,
    _Float16* __restrict__ wpack,     // [8][36][32 lanes][16 halves]
    float* __restrict__ bias)         // [128]
{
    int tid = blockIdx.x * 256 + threadIdx.x;
    if (tid >= CC * KK) return;
    int co = tid / KK;
    int k  = tid % KK;
    float inv = gamma[co] * rsqrtf(var[co] + 1e-5f);
    if (k == 0) bias[co] = beta[co] - mean[co] * inv;

    int rs = k >> 7;          // (r*3+s)
    int ci = k & 127;
    float wv = w[(co * CC + ci) * 9 + rs] * inv;

    int cot  = co >> 4;
    int m    = co & 15;
    int kc   = k >> 5;
    int kl   = k & 31;
    int lane = m + ((kl >> 4) << 4);
    int j    = kl & 15;
    wpack[(((size_t)cot * KCN + kc) * 32 + lane) * 16 + j] = (_Float16)wv;
}

// NCHW f32 spikes -> padded NHWC f16 [N][58][58][128], halo zeroed.
__global__ __launch_bounds__(256) void pack_x_kernel(
    const float* __restrict__ x, _Float16* __restrict__ x16p)
{
    int tid = blockIdx.x * 256 + threadIdx.x;
    if (tid >= NN * PPIX * CC) return;
    int c    = tid & 127;
    int rest = tid >> 7;
    int pp   = rest % PPIX;
    int n    = rest / PPIX;
    int ph   = pp / WP;
    int pw   = pp % WP;
    float v = 0.0f;
    if (ph >= 1 && ph <= HH && pw >= 1 && pw <= WW)
        v = x[((size_t)(n * CC + c)) * HWW + (ph - 1) * WW + (pw - 1)];
    x16p[tid] = (_Float16)v;   // tid == ((n*PPIX + pp)<<7) + c
}

// Zero an f16 buffer (16B per thread) — clears s1's halo before conv2.
__global__ __launch_bounds__(256) void clear_f16_kernel(float4* __restrict__ b, int n16)
{
    int tid = blockIdx.x * 256 + threadIdx.x;
    if (tid < n16) b[tid] = make_float4(0.f, 0.f, 0.f, 0.f);
}

// ---------------------------------------------------------------------------
// Implicit-GEMM 3x3 SAME conv via v_wmma_f32_16x16x32_f16 on a zero-padded
// NHWC f16 input.  Fully unrolled K loop (36 chunks), double-buffered LDS
// B-tiles (64 px x 32 ci) filled with global_load_async_to_lds_b128
// (ASYNCcnt software pipeline, one chunk ahead).
// Workgroup = 8 waves: wave w -> Co tile w (16 rows) x 64 pixels (4 accums).
// ---------------------------------------------------------------------------
__global__ __launch_bounds__(256) void conv3x3_wmma_kernel(
    const _Float16* __restrict__ in16,   // [N][58][58][128] f16, halo = 0
    const _Float16* __restrict__ wpack,  // packed A fragments
    const float*    __restrict__ bias,   // [128]
    float*          __restrict__ y)      // [N][3136][128] f32 NHWC
{
    __shared__ _Float16 ldsB[2][PTILE * 32];  // 2 x 4KB [pixel][ci] tiles

    const int n    = blockIdx.y;
    const int p0   = blockIdx.x * PTILE;
    const int tid  = threadIdx.x;
    const int wave = tid >> 5;             // co tile 0..7
    const int lane = tid & 31;

    // --- staging role: 4 threads per pixel, 16 bytes (8 halves) each -------
    const int pixB  = tid >> 2;            // 0..63
    const int lane4 = tid & 3;
    const int p  = p0 + pixB;
    const int oh = p / WW;
    const int ow = p % WW;
    // base global byte address of this thread's slice at (r=0,s=0,cic=0):
    const char* gthread = (const char*)in16 +
        (((size_t)n * PPIX + (size_t)oh * WP + ow) << 8) + (lane4 << 4);
    // LDS byte offsets (low 32 bits of generic shared pointer = LDS address)
    const unsigned ldsdst0 =
        (unsigned)(uintptr_t)(&ldsB[0][0]) + (pixB << 6) + (lane4 << 4);
    const unsigned ldsdst1 = ldsdst0 + 4096;

    // --- compute role ------------------------------------------------------
    const int col = lane & 15;             // D column (pixel within 16-group)
    const int kh  = (lane >> 4) << 4;      // B K-group per lane (0 or 16)
    const _Float16* abase = wpack + (size_t)wave * KCN * 512 + (lane << 4);
    const _Float16* bbase = &ldsB[0][(col << 5) + kh];

    v8f acc0 = {}, acc1 = {}, acc2 = {}, acc3 = {};

    // prologue: kick off chunk 0 into buffer 0
    asm volatile("global_load_async_to_lds_b128 %0, %1, off"
                 :: "v"(ldsdst0), "v"(gthread) : "memory");

#pragma unroll
    for (int kc = 0; kc < KCN; ++kc) {
        // our chunk-kc async load must have landed; then workgroup barrier
#if __has_builtin(__builtin_amdgcn_s_wait_asynccnt)
        __builtin_amdgcn_s_wait_asynccnt(0);
#else
        asm volatile("s_wait_asynccnt 0" ::: "memory");
#endif
        __syncthreads();

        // software pipeline: issue chunk kc+1 into the other buffer
        if (kc + 1 < KCN) {
            const int kn   = kc + 1;
            const int rs   = kn >> 2;                         // folds: literal
            const int goff = ((rs / 3) * WP + (rs % 3)) * 256 + (kn & 3) * 64;
            asm volatile("global_load_async_to_lds_b128 %0, %1, off"
                         :: "v"((kn & 1) ? ldsdst1 : ldsdst0),
                            "v"(gthread + goff) : "memory");
        }

        // compute on buffer kc&1  (all offsets constant-folded)
        const v16h a = *(const v16h*)(abase + kc * 512);
        const _Float16* bb = bbase + ((kc & 1) ? 2048 : 0);
        const v16h b0 = *(const v16h*)(bb);
        const v16h b1 = *(const v16h*)(bb + 512);             // pixels 16..31
        const v16h b2 = *(const v16h*)(bb + 1024);            // pixels 32..47
        const v16h b3 = *(const v16h*)(bb + 1536);            // pixels 48..63
        acc0 = __builtin_amdgcn_wmma_f32_16x16x32_f16(false, a, false, b0,
                                                      (short)0, acc0, false, false);
        acc1 = __builtin_amdgcn_wmma_f32_16x16x32_f16(false, a, false, b1,
                                                      (short)0, acc1, false, false);
        acc2 = __builtin_amdgcn_wmma_f32_16x16x32_f16(false, a, false, b2,
                                                      (short)0, acc2, false, false);
        acc3 = __builtin_amdgcn_wmma_f32_16x16x32_f16(false, a, false, b3,
                                                      (short)0, acc3, false, false);
    }

    // C/D layout: vgpr r, lane L -> D[m][ncol], m = r + 8*(L>=16), ncol = L%16
    const int co_base = (wave << 4) + ((lane >> 4) << 3);     // 8 consecutive co
    const float4 bl = *(const float4*)&bias[co_base];
    const float4 bh = *(const float4*)&bias[co_base + 4];

    float* y0 = &y[(((size_t)n * HWW + p0 + col) << 7) + co_base];
    *(float4*)(y0)     = make_float4(acc0[0] + bl.x, acc0[1] + bl.y,
                                     acc0[2] + bl.z, acc0[3] + bl.w);
    *(float4*)(y0 + 4) = make_float4(acc0[4] + bh.x, acc0[5] + bh.y,
                                     acc0[6] + bh.z, acc0[7] + bh.w);

    float* y1 = &y[(((size_t)n * HWW + p0 + 16 + col) << 7) + co_base];
    *(float4*)(y1)     = make_float4(acc1[0] + bl.x, acc1[1] + bl.y,
                                     acc1[2] + bl.z, acc1[3] + bl.w);
    *(float4*)(y1 + 4) = make_float4(acc1[4] + bh.x, acc1[5] + bh.y,
                                     acc1[6] + bh.z, acc1[7] + bh.w);

    float* y2 = &y[(((size_t)n * HWW + p0 + 32 + col) << 7) + co_base];
    *(float4*)(y2)     = make_float4(acc2[0] + bl.x, acc2[1] + bl.y,
                                     acc2[2] + bl.z, acc2[3] + bl.w);
    *(float4*)(y2 + 4) = make_float4(acc2[4] + bh.x, acc2[5] + bh.y,
                                     acc2[6] + bh.z, acc2[7] + bh.w);

    float* y3 = &y[(((size_t)n * HWW + p0 + 48 + col) << 7) + co_base];
    *(float4*)(y3)     = make_float4(acc3[0] + bl.x, acc3[1] + bl.y,
                                     acc3[2] + bl.z, acc3[3] + bl.w);
    *(float4*)(y3 + 4) = make_float4(acc3[4] + bh.x, acc3[5] + bh.y,
                                     acc3[6] + bh.z, acc3[7] + bh.w);
}

// ---------------------------------------------------------------------------
// Parametric-LIF scan over T=4: v += (x - v)*sigmoid(w); spike at v>=1;
// hard reset.  mode 0: spikes -> padded NHWC f16 (conv2 input).
// mode 1: out = s + x residual, NCHW f32 (final output).
// ---------------------------------------------------------------------------
__global__ __launch_bounds__(256) void plif_kernel(
    const float* __restrict__ y,        // [N][3136][128] f32 NHWC
    const float* __restrict__ wparam,   // scalar
    const float* __restrict__ x_nchw,   // residual (mode 1)
    float*       __restrict__ out_nchw, // final output (mode 1)
    _Float16*    __restrict__ s16p,     // padded spikes NHWC f16 (mode 0)
    int mode)
{
    int tid = blockIdx.x * 256 + threadIdx.x;
    if (tid >= BB * HWW * CC) return;
    int c    = tid & 127;
    int rest = tid >> 7;
    int pix  = rest % HWW;
    int b    = rest / HWW;
    int h    = pix / WW;
    int wcol = pix % WW;

    float kfac = 1.0f / (1.0f + __expf(-wparam[0]));
    float v = 0.0f;
    for (int t = 0; t < TT; ++t) {
        size_t off = (((size_t)(t * BB + b) * HWW + pix) << 7) + c;
        float xin = y[off];
        v += (xin - v) * kfac;
        float s = (v >= 1.0f) ? 1.0f : 0.0f;
        v *= (1.0f - s);                       // hard reset (detach)
        if (mode == 0) {
            size_t so = (((size_t)(t * BB + b) * PPIX
                          + (size_t)(h + 1) * WP + (wcol + 1)) << 7) + c;
            s16p[so] = (_Float16)s;
        } else {
            size_t o2 = ((size_t)(t * BB + b) * CC + c) * HWW + pix;
            out_nchw[o2] = s + x_nchw[o2];     // SEW 'ADD'
        }
    }
}

extern "C" void kernel_launch(void* const* d_in, const int* in_sizes, int n_in,
                              void* d_out, int out_size, void* d_ws, size_t ws_size,
                              hipStream_t stream) {
    (void)in_sizes; (void)n_in; (void)out_size; (void)ws_size;

    const float* x    = (const float*)d_in[0];
    const float* w1c  = (const float*)d_in[1];
    const float* g1   = (const float*)d_in[2];
    const float* be1  = (const float*)d_in[3];
    const float* m1   = (const float*)d_in[4];
    const float* v1   = (const float*)d_in[5];
    const float* tau1 = (const float*)d_in[6];
    const float* w2c  = (const float*)d_in[7];
    const float* g2   = (const float*)d_in[8];
    const float* be2  = (const float*)d_in[9];
    const float* m2   = (const float*)d_in[10];
    const float* v2   = (const float*)d_in[11];
    const float* tau2 = (const float*)d_in[12];
    float* out = (float*)d_out;

    char* ws = (char*)d_ws;
    size_t off = 0;
    auto bump = [&](size_t bytes) {
        void* ptr = ws + off;
        off = (off + bytes + 255) & ~(size_t)255;
        return ptr;
    };
    const size_t NPAD = (size_t)NN * PPIX * CC;          // 13,778,944 (padded)
    const size_t NOUT = (size_t)NN * HWW * CC;           // 12,845,056
    _Float16* wp1  = (_Float16*)bump((size_t)CC * KK * 2);
    _Float16* wp2  = (_Float16*)bump((size_t)CC * KK * 2);
    float*    b1   = (float*)bump(CC * 4);
    float*    b2   = (float*)bump(CC * 4);
    _Float16* x16p = (_Float16*)bump(NPAD * 2);
    float*    ybuf = (float*)bump(NOUT * 4);
    _Float16* s1p  = (_Float16*)bump(NPAD * 2);

    const int packWBlocks = (CC * KK + 255) / 256;       // 576
    pack_w_kernel<<<packWBlocks, 256, 0, stream>>>(w1c, g1, be1, m1, v1, wp1, b1);
    pack_w_kernel<<<packWBlocks, 256, 0, stream>>>(w2c, g2, be2, m2, v2, wp2, b2);

    pack_x_kernel<<<(int)((NPAD + 255) / 256), 256, 0, stream>>>(x, x16p);

    // clear s1 (halo must be zero; interior overwritten by plif mode 0)
    const int n16 = (int)(NPAD * 2 / 16);
    clear_f16_kernel<<<(n16 + 255) / 256, 256, 0, stream>>>((float4*)s1p, n16);

    dim3 convGrid(HWW / PTILE, NN);                      // 49 x 32 workgroups
    const int plifBlocks = (BB * HWW * CC + 255) / 256;  // 12544

    conv3x3_wmma_kernel<<<convGrid, 256, 0, stream>>>(x16p, wp1, b1, ybuf);
    plif_kernel<<<plifBlocks, 256, 0, stream>>>(ybuf, tau1, nullptr, nullptr, s1p, 0);
    conv3x3_wmma_kernel<<<convGrid, 256, 0, stream>>>(s1p, wp2, b2, ybuf);
    plif_kernel<<<plifBlocks, 256, 0, stream>>>(ybuf, tau2, x, out, nullptr, 1);
}
